// DimeNetStudent_36661840839109
// MI455X (gfx1250) — matrix operand order repaired and checked by reading, not compiled, not run on hardware
//
#include <hip/hip_runtime.h>
#include <stdint.h>

#define HID   128
#define NSPH  7
#define NRAD  6
#define NBIL  8
#define TPE   22           // triplets per target edge (n_atoms-2 = 22)
#define SBF_K 42           // NSPH*NRAD
#define BILK  1024         // NBIL*HID

typedef __attribute__((ext_vector_type(16))) __bf16 v16bf;
typedef __attribute__((ext_vector_type(8)))  float  v8f;

__device__ __forceinline__ float siluf(float v) {
  return v / (1.0f + expf(-v));
}

// LDS swizzle so a lane-half's 16 A values (K map {0..7,16..23} / {8..15,24..31})
// are contiguous: p(k) = (k&7) | ((k&16)>>1) | ((k&8)<<1)
__device__ __forceinline__ int aswz(int k) {
  return (k & 7) | ((k & 16) >> 1) | ((k & 8) << 1);
}

// ---------------------------------------------------------------------------
// Spherical Bessel zeros + norms, computed in double precision on device,
// replicating the reference's _jn/_bisect/_bessel_zeros exactly.
// ---------------------------------------------------------------------------
__device__ double jn_d(int l, double x) {
  double j0 = sin(x) / x;
  if (l == 0) return j0;
  double j1 = sin(x) / (x * x) - cos(x) / x;
  double jm = j0, jc = j1;
  for (int m = 1; m < l; ++m) {
    double t = (2.0 * m + 1.0) / x * jc - jm;
    jm = jc; jc = t;
  }
  return jc;
}

__global__ void k_init_tables(float* zln, float* nrm) {
  if (threadIdx.x != 0 || blockIdx.x != 0) return;
  const double PI = 3.14159265358979323846;
  double zeros[NSPH][NRAD];
  double points[NRAD + NSPH];   // 12 initial points
  for (int q = 0; q < NRAD; ++q) zeros[0][q] = (q + 1) * PI;
  int np = NRAD + NSPH - 1;     // 12
  for (int q = 0; q < np; ++q) points[q] = (q + 1) * PI;
  for (int l = 1; l < NSPH; ++l) {
    double pts[NRAD + NSPH];
    for (int q = 0; q < np - 1; ++q) {
      double a = points[q], b = points[q + 1];
      double fa = jn_d(l, a);
      for (int it = 0; it < 100; ++it) {
        double m = 0.5 * (a + b);
        double fm = jn_d(l, m);
        if (fa * fm <= 0.0) { b = m; } else { a = m; fa = fm; }
      }
      pts[q] = 0.5 * (a + b);
    }
    np -= 1;
    for (int q = 0; q < np; ++q) points[q] = pts[q];
    for (int q = 0; q < NRAD; ++q) zeros[l][q] = points[q];
  }
  for (int l = 0; l < NSPH; ++l)
    for (int n = 0; n < NRAD; ++n) {
      zln[l * NRAD + n] = (float)zeros[l][n];
      nrm[l * NRAD + n] = (float)(sqrt(2.0) / fabs(jn_d(l + 1, zeros[l][n])));
    }
}

// ---------------------------------------------------------------------------
// Per-edge: distance, envelope, radial basis (rbf) and spherical Bessel (sb)
// ---------------------------------------------------------------------------
__global__ void k_edge(const float* __restrict__ pos,
                       const int* __restrict__ esrc, const int* __restrict__ edst,
                       const float* __restrict__ freq,
                       const float* __restrict__ zln, const float* __restrict__ nrm,
                       float* __restrict__ rbf, float* __restrict__ sb, int E) {
  int e = blockIdx.x * blockDim.x + threadIdx.x;
  if (e >= E) return;
  int i = edst[e], j = esrc[e];
  float dx = pos[i * 3 + 0] - pos[j * 3 + 0];
  float dy = pos[i * 3 + 1] - pos[j * 3 + 1];
  float dz = pos[i * 3 + 2] - pos[j * 3 + 2];
  float dist = sqrtf(dx * dx + dy * dy + dz * dz);
  float d = dist * (1.0f / 5.0f);
  float x5 = d * d * d * d * d;
  float env = (d < 1.0f)
                  ? (1.0f / d - 28.0f * x5 + 48.0f * x5 * d - 21.0f * x5 * d * d)
                  : 0.0f;
  for (int n = 0; n < NRAD; ++n)
    rbf[(size_t)e * NRAD + n] = env * sinf(freq[n] * d);
  for (int l = 0; l < NSPH; ++l) {
    for (int n = 0; n < NRAD; ++n) {
      float xx = d * zln[l * NRAD + n];
      float j0 = sinf(xx) / xx;
      float v;
      if (l == 0) v = j0;
      else {
        float j1 = sinf(xx) / (xx * xx) - cosf(xx) / xx;
        float jm = j0, jc = j1;
        for (int m = 1; m < l; ++m) {
          float t = (2.0f * m + 1.0f) / xx * jc - jm;
          jm = jc; jc = t;
        }
        v = jc;
      }
      sb[(size_t)e * SBF_K + l * NRAD + n] = nrm[l * NRAD + n] * env * v;
    }
  }
}

// ---------------------------------------------------------------------------
// Per-triplet Legendre basis. cos(atan2(|cross|,dot)) == dot/||.|| directly.
// ---------------------------------------------------------------------------
__global__ void k_cbf(const float* __restrict__ pos,
                      const int* __restrict__ esrc, const int* __restrict__ edst,
                      const int* __restrict__ idx_kj, const int* __restrict__ idx_ji,
                      float* __restrict__ cbf, int T) {
  int t = blockIdx.x * blockDim.x + threadIdx.x;
  if (t >= T) return;
  int e1 = idx_ji[t];
  int ii = edst[e1], jj = esrc[e1];
  int kk = esrc[idx_kj[t]];
  float ax = pos[jj * 3 + 0] - pos[ii * 3 + 0];
  float ay = pos[jj * 3 + 1] - pos[ii * 3 + 1];
  float az = pos[jj * 3 + 2] - pos[ii * 3 + 2];
  float bx = pos[kk * 3 + 0] - pos[ii * 3 + 0];
  float by = pos[kk * 3 + 1] - pos[ii * 3 + 1];
  float bz = pos[kk * 3 + 2] - pos[ii * 3 + 2];
  float dot = ax * bx + ay * by + az * bz;
  float cx = ay * bz - az * by, cy = az * bx - ax * bz, cz = ax * by - ay * bx;
  float crs2 = cx * cx + cy * cy + cz * cz;
  float c = dot * rsqrtf(dot * dot + crs2);
  float P[NSPH];
  P[0] = 1.0f; P[1] = c;
  for (int l = 1; l < NSPH - 1; ++l)
    P[l + 1] = ((2.0f * l + 1.0f) * c * P[l] - (float)l * P[l - 1]) / (float)(l + 1);
  const float INV4PI = 0.07957747154594767f;  // 1/(4*pi)
#pragma unroll
  for (int l = 0; l < NSPH; ++l)
    cbf[(size_t)t * NSPH + l] = P[l] * sqrtf((2.0f * l + 1.0f) * INV4PI);
}

// ---------------------------------------------------------------------------
// Embedding concat rows: [h_z[dst] | h_z[src] | silu(rbf@rbf_w+rbf_b)]
// ---------------------------------------------------------------------------
__global__ void k_emb_concat(const float* __restrict__ embw, const int* __restrict__ z,
                             const int* __restrict__ esrc, const int* __restrict__ edst,
                             const float* __restrict__ rbf,
                             const float* __restrict__ rbfw, const float* __restrict__ rbfb,
                             float* __restrict__ Aemb, int E) {
  int idx = blockIdx.x * blockDim.x + threadIdx.x;
  if (idx >= E * 384) return;
  int e = idx / 384, c = idx % 384;
  float v;
  if (c < 128) {
    v = embw[(size_t)z[edst[e]] * HID + c];
  } else if (c < 256) {
    v = embw[(size_t)z[esrc[e]] * HID + (c - 128)];
  } else {
    int hh = c - 256;
    float acc = rbfb[hh];
    for (int n = 0; n < NRAD; ++n)
      acc += rbf[(size_t)e * NRAD + n] * rbfw[n * HID + hh];
    v = siluf(acc);
  }
  Aemb[(size_t)e * 384 + c] = v;
}

// ---------------------------------------------------------------------------
// Weight converts: f32 [K x 128] -> bf16 in WMMA B-fragment order:
//   Bf[(kb*128 + col)*32 + kin]  =  B[(kb*32 + kin)*128 + col]
// so each lane's 16-element fragment is one contiguous 32-byte global load.
// ---------------------------------------------------------------------------
__global__ void k_cvt_frag(const float* __restrict__ s, __bf16* __restrict__ d, int K) {
  int i = blockIdx.x * blockDim.x + threadIdx.x;
  if (i >= K * HID) return;
  int kin = i & 31, c = (i >> 5) & 127, kb = i >> 12;
  d[i] = (__bf16)s[(size_t)(kb * 32 + kin) * HID + c];
}

// interaction W [o=128][b=8][h=128] -> fragment-order B with k = b*128+h, col = o
__global__ void k_cvt_W_frag(const float* __restrict__ W, __bf16* __restrict__ d) {
  int i = blockIdx.x * blockDim.x + threadIdx.x;
  if (i >= BILK * HID) return;
  int kin = i & 31, c = (i >> 5) & 127, kb = i >> 12;
  d[i] = (__bf16)W[(size_t)c * BILK + kb * 32 + kin];
}

// ---------------------------------------------------------------------------
// WMMA GEMM:  C[M x 128] = epilogue( A[M x K] @ B[K x 128] )
// A: f32 (ABF16=0) or bf16 (ABF16=1), staged row-major (no swizzle) in LDS.
// B: bf16, pre-arranged in fragment order, loaded straight from global (L2).
// Workgroup = 256 threads = 8 waves; wave w owns output columns 16w..16w+15.
// K must be a multiple of 64. epilogue: out = (add? add : 0) + act(acc + bias)
// ---------------------------------------------------------------------------
template <int ABF16>
__global__ void k_gemm(const void* __restrict__ Aptr,
                       const __bf16* __restrict__ Bw,
                       const float* __restrict__ bias,
                       const float* __restrict__ add,
                       float* __restrict__ Cout, int M, int K, int act) {
  __shared__ alignas(16) __bf16 As[16][64];   // two 32-K blocks, row-major, unswizzled
  int tid = threadIdx.x;
  int wave = tid >> 5, lane = tid & 31;
  int m0 = blockIdx.x * 16;
  const float*  Af = (const float*)Aptr;
  const __bf16* Ab = (const __bf16*)Aptr;
  int row  = lane & 15;
  int half = (lane >= 16) ? 16 : 0;
  int col  = wave * 16 + (lane & 15);
  v8f acc = {0.f, 0.f, 0.f, 0.f, 0.f, 0.f, 0.f, 0.f};

  for (int kk = 0; kk < K; kk += 64) {
    // stage A tile (16x64), 4 elements / thread, row-major, no swizzle
#pragma unroll
    for (int u = 0; u < 4; ++u) {
      int idx = tid * 4 + u;                 // 0..1023
      int r = idx >> 6, c = idx & 63;        // row, k-in-tile
      float v = ABF16 ? (float)Ab[(size_t)(m0 + r) * K + kk + c]
                      : Af[(size_t)(m0 + r) * K + kk + c];
      As[r][c] = (__bf16)v;
    }
    // prefetch next A tile (gfx1250 global_prefetch path)
    if (kk + 64 < K) {
      int idx = tid * 4;
      int r = idx >> 6, c = idx & 63;
      if (ABF16) __builtin_prefetch(&Ab[(size_t)(m0 + r) * K + kk + 64 + c], 0, 0);
      else       __builtin_prefetch(&Af[(size_t)(m0 + r) * K + kk + 64 + c], 0, 0);
    }
    __syncthreads();

#pragma unroll
    for (int s = 0; s < 2; ++s) {
      // A fragment: one contiguous 32B LDS read (ds_load_b128 x2)
      v16bf a = *reinterpret_cast<const v16bf*>(&As[row][s * 32 + half]);
      // B fragment: one contiguous 32B global read (global_load_b128 x2)
      size_t kb = (size_t)((kk >> 5) + s);
      v16bf b = *reinterpret_cast<const v16bf*>(Bw + (kb * HID + col) * 32 + half);
      acc = __builtin_amdgcn_wmma_f32_16x16x32_bf16(
          /*neg_a=*/false, a, /*neg_b=*/false, b,
          /*c_mod=*/(short)0, acc, /*reuse_a=*/false, /*reuse_b=*/false);
    }
    __syncthreads();
  }

  // epilogue: C/D layout — VGPR r: M = r (lanes 0-15) / 8+r (lanes 16-31)
  int rbase = (lane >= 16) ? 8 : 0;
#pragma unroll
  for (int r = 0; r < 8; ++r) {
    int orow = m0 + rbase + r;
    float v = acc[r];
    if (bias) v += bias[col];
    if (act)  v = siluf(v);
    if (add)  v += add[(size_t)orow * HID + col];
    Cout[(size_t)orow * HID + col] = v;
  }
}

// ---------------------------------------------------------------------------
// x_kj *= (rbf @ rbf_w)   (in place, per (edge, channel))
// ---------------------------------------------------------------------------
__global__ void k_xkj_scale(float* __restrict__ g, const float* __restrict__ rbf,
                            const float* __restrict__ rbfw, int E) {
  int idx = blockIdx.x * blockDim.x + threadIdx.x;
  if (idx >= E * HID) return;
  int e = idx / HID, hh = idx % HID;
  float acc = 0.f;
  for (int n = 0; n < NRAD; ++n)
    acc += rbf[(size_t)e * NRAD + n] * rbfw[n * HID + hh];
  g[idx] *= acc;
}

// ---------------------------------------------------------------------------
// Per target edge e: S[e, b*128+h] = sum_{r<22} sbf_p[r,b] * x_kj[idx_kj, h]
// with sbf_p[r,b] = sum_{l,n} sb[idx_kj, ln] * cbf[t, l] * sbf_w[ln, b].
// This is the factorized einsum + segment_sum (idx_ji is contiguous: t=e*22+r)
// ---------------------------------------------------------------------------
__global__ void k_build_S(const float* __restrict__ xkj, const float* __restrict__ sb,
                          const float* __restrict__ cbf, const float* __restrict__ sbf_w,
                          const int* __restrict__ idx_kj, __bf16* __restrict__ S, int E) {
  __shared__ float xk[TPE][HID];
  __shared__ float sbr[TPE][SBF_K];
  __shared__ float cb[TPE][NSPH];
  __shared__ float sw[SBF_K * NBIL];
  __shared__ float sp[TPE][NBIL];
  int e = blockIdx.x;
  int tid = threadIdx.x;           // 0..127
  int t0 = e * TPE;
  for (int u = tid; u < SBF_K * NBIL; u += 128) sw[u] = sbf_w[u];
  for (int r = 0; r < TPE; ++r) {
    int src = idx_kj[t0 + r];
    xk[r][tid] = xkj[(size_t)src * HID + tid];
    if (tid < SBF_K) sbr[r][tid] = sb[(size_t)src * SBF_K + tid];
    if (tid < NSPH)  cb[r][tid]  = cbf[(size_t)(t0 + r) * NSPH + tid];
  }
  __syncthreads();
  for (int u = tid; u < TPE * NBIL; u += 128) {
    int r = u >> 3, b = u & 7;
    float acc = 0.f;
    for (int l = 0; l < NSPH; ++l) {
      float cl = cb[r][l];
      for (int n = 0; n < NRAD; ++n)
        acc += sbr[r][l * NRAD + n] * cl * sw[(l * NRAD + n) * NBIL + b];
    }
    sp[r][b] = acc;
  }
  __syncthreads();
  for (int b = 0; b < NBIL; ++b) {
    float acc = 0.f;
#pragma unroll
    for (int r = 0; r < TPE; ++r) acc += sp[r][b] * xk[r][tid];
    S[(size_t)e * BILK + b * HID + tid] = (__bf16)acc;
  }
}

// ---------------------------------------------------------------------------
// Output block scatter: OT[dst[e], h] += x[e,h] * (rbf @ rbf_w)[e,h]
// ---------------------------------------------------------------------------
__global__ void k_scatter_out(const float* __restrict__ x, const float* __restrict__ rbf,
                              const float* __restrict__ rbfw, const int* __restrict__ edst,
                              float* __restrict__ OT, int E) {
  int idx = blockIdx.x * blockDim.x + threadIdx.x;
  if (idx >= E * HID) return;
  int e = idx / HID, hh = idx % HID;
  float acc = 0.f;
  for (int n = 0; n < NRAD; ++n)
    acc += rbf[(size_t)e * NRAD + n] * rbfw[n * HID + hh];
  atomicAdd(&OT[(size_t)edst[e] * HID + hh], x[idx] * acc);
}

__global__ void k_pnode(const float* __restrict__ t3, const float* __restrict__ outw,
                        float* __restrict__ pnode, int NN) {
  int n = blockIdx.x * blockDim.x + threadIdx.x;
  if (n >= NN) return;
  float acc = 0.f;
  for (int hh = 0; hh < HID; ++hh) acc += t3[(size_t)n * HID + hh] * outw[hh];
  pnode[n] += acc;
}

__global__ void k_graph_sum(const float* __restrict__ pnode, const int* __restrict__ batch,
                            float* __restrict__ out, int NN) {
  int n = blockIdx.x * blockDim.x + threadIdx.x;
  if (n >= NN) return;
  atomicAdd(&out[batch[n]], pnode[n]);
}

// ---------------------------------------------------------------------------
// Host orchestration
// ---------------------------------------------------------------------------
extern "C" void kernel_launch(void* const* d_in, const int* in_sizes, int n_in,
                              void* d_out, int out_size, void* d_ws, size_t ws_size,
                              hipStream_t stream) {
  (void)ws_size;
  // ---- param leaf table (jax tree_flatten order: sorted dict keys, DFS) ----
  int lsz[101]; int p = 0;
  lsz[p++] = 95 * HID; lsz[p++] = HID; lsz[p++] = 384 * HID; lsz[p++] = HID;
  lsz[p++] = NRAD * HID; lsz[p++] = NRAD;
  for (int k = 0; k < 3; ++k) {
    const int a[21] = {HID * NBIL * HID, HID, HID, HID * HID, HID * HID,
                       HID, HID, HID * HID, HID * HID,
                       HID, HID, HID * HID, HID * HID,
                       HID, HID * HID, HID, HID * HID, HID, HID * HID,
                       NRAD * HID, SBF_K * NBIL};
    for (int q = 0; q < 21; ++q) lsz[p++] = a[q];
  }
  for (int k = 0; k < 4; ++k) {
    const int a[8] = {HID, HID * HID, HID, HID * HID, HID, HID * HID, HID, NRAD * HID};
    for (int q = 0; q < 8; ++q) lsz[p++] = a[q];
  }
  long long loff[101]; long long accum = 0;
  for (int q = 0; q < 101; ++q) { loff[q] = accum; accum += lsz[q]; }

  const bool leafwise = (n_in > 20);
  const float* packed = leafwise ? nullptr : (const float*)d_in[1];
  auto LF = [&](int k) -> const float* {
    return leafwise ? (const float*)d_in[1 + k] : packed + loff[k];
  };
  const int base = leafwise ? 102 : 2;
  const float* pos  = (const float*)d_in[0];
  const int* z      = (const int*)d_in[base + 0];
  const int* batch  = (const int*)d_in[base + 1];
  const int* esrc   = (const int*)d_in[base + 2];
  const int* edst   = (const int*)d_in[base + 3];
  const int* ikj    = (const int*)d_in[base + 4];
  const int* iji    = (const int*)d_in[base + 5];
  const int NN = in_sizes[0] / 3;
  const int E  = in_sizes[base + 2];
  const int T  = in_sizes[base + 4];
  const int B  = out_size;

  // ---- workspace ----
  char* wp = (char*)d_ws;
  auto alloc = [&](size_t bytes) -> void* {
    void* q = (void*)wp; wp += (bytes + 255) & ~(size_t)255; return q;
  };
  float* zln  = (float*)alloc(SBF_K * 4);
  float* nrmv = (float*)alloc(SBF_K * 4);
  float* rbf  = (float*)alloc((size_t)E * NRAD * 4);
  float* sb   = (float*)alloc((size_t)E * SBF_K * 4);
  float* cbf  = (float*)alloc((size_t)T * NSPH * 4);
  float* x    = (float*)alloc((size_t)E * HID * 4);
  float* xn   = (float*)alloc((size_t)E * HID * 4);
  float* xji  = (float*)alloc((size_t)E * HID * 4);
  float* g    = (float*)alloc((size_t)E * HID * 4);
  float* hbuf = (float*)alloc((size_t)E * HID * 4);
  float* t1   = (float*)alloc((size_t)E * HID * 4);
  __bf16* S   = (__bf16*)alloc((size_t)E * BILK * 2);  // reused as f32 Aemb (E*384*4 < E*1024*2)
  __bf16* bw  = (__bf16*)alloc((size_t)BILK * HID * 2);
  float* OT   = (float*)alloc((size_t)NN * HID * 4);
  float* O1   = (float*)alloc((size_t)NN * HID * 4);
  float* O2   = (float*)alloc((size_t)NN * HID * 4);
  float* pnode = (float*)alloc((size_t)NN * 4);

  auto cvt = [&](const float* src, int K) {
    k_cvt_frag<<<(K * HID + 255) / 256, 256, 0, stream>>>(src, bw, K);
  };
  auto gemm = [&](const void* A, int a_bf16, const float* bias, const float* add,
                  float* C, int M, int K, int act) {
    if (a_bf16)
      k_gemm<1><<<M / 16, 256, 0, stream>>>(A, bw, bias, add, C, M, K, act);
    else
      k_gemm<0><<<M / 16, 256, 0, stream>>>(A, bw, bias, add, C, M, K, act);
  };

  // ---- bases ----
  k_init_tables<<<1, 1, 0, stream>>>(zln, nrmv);
  k_edge<<<(E + 127) / 128, 128, 0, stream>>>(pos, esrc, edst, LF(5), zln, nrmv, rbf, sb, E);
  k_cbf<<<(T + 255) / 256, 256, 0, stream>>>(pos, esrc, edst, ikj, iji, cbf, T);

  // ---- embedding ----
  float* Aemb = (float*)S;
  k_emb_concat<<<(unsigned)(((size_t)E * 384 + 255) / 256), 256, 0, stream>>>(
      LF(0), z, esrc, edst, rbf, LF(4), LF(3), Aemb, E);
  cvt(LF(2), 384);
  gemm(Aemb, 0, LF(1), nullptr, x, E, 384, 1);

  (void)hipMemsetAsync(pnode, 0, (size_t)NN * 4, stream);
  (void)hipMemsetAsync(d_out, 0, (size_t)B * 4, stream);

  auto out_block = [&](int ob, const float* xcur) {
    int b0 = 69 + ob * 8;
    (void)hipMemsetAsync(OT, 0, (size_t)NN * HID * 4, stream);
    k_scatter_out<<<(unsigned)(((size_t)E * HID + 255) / 256), 256, 0, stream>>>(
        xcur, rbf, LF(b0 + 7), edst, OT, E);
    cvt(LF(b0 + 1), HID); gemm(OT, 0, LF(b0 + 0), nullptr, O1, NN, HID, 1);
    cvt(LF(b0 + 3), HID); gemm(O1, 0, LF(b0 + 2), nullptr, O2, NN, HID, 1);
    cvt(LF(b0 + 5), HID); gemm(O2, 0, LF(b0 + 4), nullptr, O1, NN, HID, 1);
    k_pnode<<<(NN + 255) / 256, 256, 0, stream>>>(O1, LF(b0 + 6), pnode, NN);
  };

  out_block(0, x);

  float* xcur = x; float* xnext = xn;
  for (int k = 0; k < 3; ++k) {
    int ib = 6 + 21 * k;
    // x_ji, x_kj
    cvt(LF(ib + 14), HID); gemm(xcur, 0, LF(ib + 13), nullptr, xji, E, HID, 1);
    cvt(LF(ib + 16), HID); gemm(xcur, 0, LF(ib + 15), nullptr, g,   E, HID, 1);
    k_xkj_scale<<<(unsigned)(((size_t)E * HID + 255) / 256), 256, 0, stream>>>(
        g, rbf, LF(ib + 19), E);
    // factorized einsum + segment_sum
    k_build_S<<<E, 128, 0, stream>>>(g, sb, cbf, LF(ib + 20), ikj, S, E);
    k_cvt_W_frag<<<(BILK * HID + 255) / 256, 256, 0, stream>>>(LF(ib + 0), bw);
    gemm(S, 1, nullptr, xji, hbuf, E, BILK, 0);        // h = x_ji + S @ W2t
    // before-residual
    cvt(LF(ib + 11), HID); gemm(hbuf, 0, LF(ib + 9),  nullptr, t1,   E, HID, 1);
    cvt(LF(ib + 12), HID); gemm(t1,   0, LF(ib + 10), hbuf,    hbuf, E, HID, 1);
    // lin + skip connection to x
    cvt(LF(ib + 18), HID); gemm(hbuf, 0, LF(ib + 17), xcur, xnext, E, HID, 1);
    // after-residuals x2
    cvt(LF(ib + 3), HID); gemm(xnext, 0, LF(ib + 1), nullptr, t1,    E, HID, 1);
    cvt(LF(ib + 4), HID); gemm(t1,    0, LF(ib + 2), xnext,   xnext, E, HID, 1);
    cvt(LF(ib + 7), HID); gemm(xnext, 0, LF(ib + 5), nullptr, t1,    E, HID, 1);
    cvt(LF(ib + 8), HID); gemm(t1,    0, LF(ib + 6), xnext,   xnext, E, HID, 1);
    // swap
    float* tmp = xcur; xcur = xnext; xnext = tmp;
    out_block(k + 1, xcur);
  }

  k_graph_sum<<<(NN + 255) / 256, 256, 0, stream>>>(pnode, batch, (float*)d_out, NN);
}
